// RWKV7TimeMix_75222057222698
// MI455X (gfx1250) — compile-verified, matrix operand order checked by reading, not run on hardware
//
#include <hip/hip_runtime.h>
#include <hip/hip_bf16.h>
#include <math.h>

// ---------------------------------------------------------------------------
// RWKV7 time-mix for MI455X (gfx1250): bf16 WMMA GEMMs with double-buffered
// async LDS staging (ASYNCcnt), wave32 reductions, register-resident scan.
// ---------------------------------------------------------------------------

typedef __bf16 bf16_t;
typedef __attribute__((ext_vector_type(16))) __bf16 v16bf;
typedef __attribute__((ext_vector_type(8)))  float  v8f;

#define CC   2048
#define HEADS 32
#define HS   64
#define BB   2
#define TT   4096
#define NBT  (BB*TT)          // 8192 token rows

union FragBF { v16bf v; uint4 q[2]; };

__device__ __forceinline__ float sigmoidf_(float x) { return 1.0f / (1.0f + expf(-x)); }

__device__ __forceinline__ void wait_async0()
{
    asm volatile("s_wait_asynccnt 0" ::: "memory");
}

// ---------------------------------------------------------------------------
// Generic bf16 NT GEMM:  D[m,n] = sum_k A[m*K+k] * Wt[n*K+k]   (fp32 accum/out)
// Block tile: 256(M) x 128(N), K-step 32.  8 waves = 4 M-slots x 2 N-slots,
// each wave computes 4x4 = 16 WMMA 16x16 tiles (64 rows x 64 cols).
// Staging: GLOBAL_LOAD_ASYNC_TO_LDS_B128, double-buffered so the async engine
// fills tile k+1 while WMMAs consume tile k (one barrier per K-step).
// INST_OFFSET applies to both global and LDS addresses, so each 4/2-chunk
// burst shares one base register pair with offset: immediates.
// Requires: M % 256 == 0, K % 32 == 0.  N arbitrary (guarded).
// ---------------------------------------------------------------------------
__global__ void __launch_bounds__(256)
gemm_bf16_nt(const bf16_t* __restrict__ A, const bf16_t* __restrict__ Wt,
             float* __restrict__ D, int M, int N, int K)
{
    __shared__ alignas(16) bf16_t sA[2][256 * 32];   // 2 x 16 KB
    __shared__ alignas(16) bf16_t sB[2][128 * 32];   // 2 x  8 KB

    const int tid  = threadIdx.x;
    const int lane = tid & 31;
    const int wv   = tid >> 5;
    const int wm   = wv & 3;          // 0..3  (64 rows each)
    const int wn   = wv >> 2;         // 0..1  (64 cols each)
    const int lr   = lane & 15;
    const int hi   = lane >> 4;

    const int bm = blockIdx.y * 256;
    const int bn = blockIdx.x * 128;

    v8f acc[4][4];
#pragma unroll
    for (int i = 0; i < 4; i++)
#pragma unroll
        for (int j = 0; j < 4; j++)
#pragma unroll
            for (int e = 0; e < 8; e++) acc[i][j][e] = 0.0f;

    // staging assignment:
    //  A: thread t stages row t (32 bf16 = 4 x b128)
    //  B: thread t stages row t>>1, half (t&1) (16 bf16 = 2 x b128)
    const int brow = tid >> 1;
    const int bcol = (tid & 1) * 16;
    const int nrow = bn + brow;
    const bool nvalid = (nrow < N);

    auto stage = [&](int buf, int k0) {
        {
            const bf16_t* gA = A + (size_t)(bm + tid) * K + k0;
            const uint32_t laA = (uint32_t)(uintptr_t)&sA[buf][tid * 32];
            asm volatile(
                "global_load_async_to_lds_b128 %0, %1, off\n\t"
                "global_load_async_to_lds_b128 %0, %1, off offset:16\n\t"
                "global_load_async_to_lds_b128 %0, %1, off offset:32\n\t"
                "global_load_async_to_lds_b128 %0, %1, off offset:48"
                :: "v"(laA), "v"(gA) : "memory");
        }
        if (nvalid) {
            const bf16_t* gB = Wt + (size_t)nrow * K + k0 + bcol;
            const uint32_t laB = (uint32_t)(uintptr_t)&sB[buf][brow * 32 + bcol];
            asm volatile(
                "global_load_async_to_lds_b128 %0, %1, off\n\t"
                "global_load_async_to_lds_b128 %0, %1, off offset:16"
                :: "v"(laB), "v"(gB) : "memory");
        } else {
            const uint4 z = make_uint4(0u, 0u, 0u, 0u);
            *(uint4*)&sB[buf][brow * 32 + bcol]     = z;
            *(uint4*)&sB[buf][brow * 32 + bcol + 8] = z;
        }
    };

    const int nk = K >> 5;
    int buf = 0;
    stage(0, 0);                          // prologue: tile 0 in flight

    for (int kt = 0; kt < nk; ++kt) {
        wait_async0();                    // my writes of tile kt are in LDS
        __syncthreads();                  // everyone's are (and tile kt-1 reads done)
        if (kt + 1 < nk)
            stage(buf ^ 1, (kt + 1) * 32);   // DMA next tile during compute

        FragBF a[4], b[4];
#pragma unroll
        for (int i = 0; i < 4; i++) {
            const int row = wm * 64 + i * 16 + lr;
            a[i].q[0] = *(const uint4*)&sA[buf][row * 32 + hi * 8];
            a[i].q[1] = *(const uint4*)&sA[buf][row * 32 + 16 + hi * 8];
        }
#pragma unroll
        for (int j = 0; j < 4; j++) {
            const int row = wn * 64 + j * 16 + lr;
            b[j].q[0] = *(const uint4*)&sB[buf][row * 32 + hi * 8];
            b[j].q[1] = *(const uint4*)&sB[buf][row * 32 + 16 + hi * 8];
        }
#pragma unroll
        for (int i = 0; i < 4; i++)
#pragma unroll
            for (int j = 0; j < 4; j++)
                acc[i][j] = __builtin_amdgcn_wmma_f32_16x16x32_bf16(
                    false, a[i].v, false, b[j].v, (short)0, acc[i][j], false, false);
        buf ^= 1;
    }

    // epilogue: VGPR e -> row (e + hi*8), col = lr
#pragma unroll
    for (int i = 0; i < 4; i++) {
        const int mrow = bm + wm * 64 + i * 16 + hi * 8;
#pragma unroll
        for (int j = 0; j < 4; j++) {
            const int ncol = bn + wn * 64 + j * 16 + lr;
            if (ncol < N) {
#pragma unroll
                for (int e = 0; e < 8; e++)
                    D[(size_t)(mrow + e) * N + ncol] = acc[i][j][e];
            }
        }
    }
}

// ---------------------------------------------------------------------------
// Token shift + six interpolations -> bf16 staged activations
// ---------------------------------------------------------------------------
__global__ void __launch_bounds__(256)
prep_x(const float* __restrict__ x, const float* __restrict__ shift,
       const float* __restrict__ mr, const float* __restrict__ mw,
       const float* __restrict__ mk, const float* __restrict__ mv,
       const float* __restrict__ ma, const float* __restrict__ mg,
       bf16_t* xr, bf16_t* xw, bf16_t* xk, bf16_t* xv, bf16_t* xa, bf16_t* xg)
{
    const size_t i = (size_t)blockIdx.x * blockDim.x + threadIdx.x;
    if (i >= (size_t)NBT * CC) return;
    const int c = (int)(i % CC);
    const size_t bt = i / CC;
    const int t = (int)(bt % TT);
    const int b = (int)(bt / TT);
    const float xc = x[i];
    const float xp = t ? x[i - CC] : shift[(size_t)b * CC + c];
    const float dx = xp - xc;
    xr[i] = (bf16_t)(xc + dx * mr[c]);
    xw[i] = (bf16_t)(xc + dx * mw[c]);
    xk[i] = (bf16_t)(xc + dx * mk[c]);
    xv[i] = (bf16_t)(xc + dx * mv[c]);
    xa[i] = (bf16_t)(xc + dx * ma[c]);
    xg[i] = (bf16_t)(xc + dx * mg[c]);
}

// f32 -> bf16 with optional activation (0=copy, 1=tanh, 2=sigmoid)
__global__ void __launch_bounds__(256)
act_cvt(const float* __restrict__ in, bf16_t* __restrict__ out, size_t n, int mode)
{
    const size_t i = (size_t)blockIdx.x * blockDim.x + threadIdx.x;
    if (i >= n) return;
    float v = in[i];
    if (mode == 1) v = tanhf(v);
    else if (mode == 2) v = sigmoidf_(v);
    out[i] = (bf16_t)v;
}

// transpose-convert: in (K,N) f32  ->  out (N,K) bf16
__global__ void __launch_bounds__(256)
cvt_t(const float* __restrict__ in, bf16_t* __restrict__ out, int K, int N)
{
    const size_t i = (size_t)blockIdx.x * blockDim.x + threadIdx.x;
    if (i >= (size_t)K * N) return;
    const int n = (int)(i / K);
    const int k = (int)(i % K);
    out[(size_t)n * K + k] = (bf16_t)in[(size_t)k * N + n];
}

// ---------------------------------------------------------------------------
// post-mix: decay, iclr, value-mix, kk normalization, k update.
// One wave (32 lanes) per (b,t,head); 2 channels per lane.
// ---------------------------------------------------------------------------
__global__ void __launch_bounds__(256)
post_mix(float* __restrict__ kbuf, float* __restrict__ vbuf,
         float* __restrict__ wbuf, float* __restrict__ abuf,
         float* __restrict__ vlbuf,
         const float* __restrict__ v_first,
         const float* __restrict__ w0, const float* __restrict__ a0,
         const float* __restrict__ v0,
         const float* __restrict__ k_k, const float* __restrict__ k_a)
{
    const int lane = threadIdx.x & 31;
    const int gw = blockIdx.x * 8 + (threadIdx.x >> 5);   // wave -> (bt, h)
    const int h  = gw % HEADS;
    const size_t bt = (size_t)(gw / HEADS);
    const int c0 = h * HS + lane * 2;
    const size_t idx = bt * CC + c0;

    float kk0, kk1, kraw0, kraw1, ic0, ic1;
    {
        const float wl0 = wbuf[idx], wl1 = wbuf[idx + 1];
        wbuf[idx]     = expf(-0.606531f * sigmoidf_(w0[c0] + wl0));
        wbuf[idx + 1] = expf(-0.606531f * sigmoidf_(w0[c0 + 1] + wl1));

        ic0 = sigmoidf_(a0[c0] + abuf[idx]);
        ic1 = sigmoidf_(a0[c0 + 1] + abuf[idx + 1]);
        abuf[idx] = ic0; abuf[idx + 1] = ic1;

        const float vm0 = sigmoidf_(v0[c0] + vlbuf[idx]);
        const float vm1 = sigmoidf_(v0[c0 + 1] + vlbuf[idx + 1]);
        const float vv0 = vbuf[idx], vv1 = vbuf[idx + 1];
        vbuf[idx]     = vv0 + (v_first[idx] - vv0) * vm0;
        vbuf[idx + 1] = vv1 + (v_first[idx + 1] - vv1) * vm1;

        kraw0 = kbuf[idx]; kraw1 = kbuf[idx + 1];
        kk0 = kraw0 * k_k[c0];
        kk1 = kraw1 * k_k[c0 + 1];
    }
    // head-wide sum of squares (wave32 butterfly)
    float ss = kk0 * kk0 + kk1 * kk1;
#pragma unroll
    for (int m = 16; m >= 1; m >>= 1) ss += __shfl_xor(ss, m, 32);
    const float den = 1.0f / fmaxf(sqrtf(ss), 1e-12f);
    vlbuf[idx]     = kk0 * den;
    vlbuf[idx + 1] = kk1 * den;

    kbuf[idx]     = kraw0 * (1.0f + (ic0 - 1.0f) * k_a[c0]);
    kbuf[idx + 1] = kraw1 * (1.0f + (ic1 - 1.0f) * k_a[c0 + 1]);
}

// ---------------------------------------------------------------------------
// Sequential WKV scan. One block per (b,h); 64 threads, thread i holds state
// row s[i][0..63] in registers.  a = -kk, b = kk*iclr computed on the fly.
// ---------------------------------------------------------------------------
__global__ void __launch_bounds__(64)
rwkv_scan(const float* __restrict__ r,  const float* __restrict__ w,
          const float* __restrict__ k,  const float* __restrict__ v,
          const float* __restrict__ kk, const float* __restrict__ iclr,
          const float* __restrict__ s0, float* __restrict__ y)
{
    const int b = blockIdx.x / HEADS;
    const int h = blockIdx.x % HEADS;
    const int i = threadIdx.x;

    __shared__ float sr[HS], sw[HS], sk[HS], sv[HS], sa[HS], sb[HS];

    float s[HS];
    const float* sp = s0 + (((size_t)b * HEADS + h) * HS + i) * HS;
#pragma unroll
    for (int kx = 0; kx < HS; kx++) s[kx] = sp[kx];

    const size_t base = (size_t)b * TT * CC + (size_t)h * HS;
    for (int t = 0; t < TT; t++) {
        const size_t idx = base + (size_t)t * CC + i;
        const float kkv = kk[idx];
        sr[i] = r[idx]; sw[i] = w[idx]; sk[i] = k[idx]; sv[i] = v[idx];
        sa[i] = -kkv;   sb[i] = kkv * iclr[idx];
        __syncthreads();

        float acc_sa = 0.0f;
#pragma unroll
        for (int kx = 0; kx < HS; kx++) acc_sa += s[kx] * sa[kx];

        const float vi = sv[i];
        float yo = 0.0f;
#pragma unroll
        for (int kx = 0; kx < HS; kx++) {
            const float ns = s[kx] * sw[kx] + acc_sa * sb[kx] + vi * sk[kx];
            s[kx] = ns;
            yo += ns * sr[kx];
        }
        y[idx] = yo;
        __syncthreads();
    }
}

// ---------------------------------------------------------------------------
// GroupNorm (per head) + bonus + gate -> bf16 staged for final GEMM.
// One wave per (b,t,head); 2 channels per lane.
// ---------------------------------------------------------------------------
__global__ void __launch_bounds__(256)
gn_bonus(const float* __restrict__ att, const float* __restrict__ r,
         const float* __restrict__ k,   const float* __restrict__ v,
         const float* __restrict__ g,   const float* __restrict__ ln_w,
         const float* __restrict__ ln_b, const float* __restrict__ r_k,
         bf16_t* __restrict__ gated)
{
    const int lane = threadIdx.x & 31;
    const int gw = blockIdx.x * 8 + (threadIdx.x >> 5);
    const int h  = gw % HEADS;
    const size_t bt = (size_t)(gw / HEADS);
    const int c0 = h * HS + lane * 2;
    const size_t idx = bt * CC + c0;

    const float a0_ = att[idx], a1_ = att[idx + 1];
    float sum = a0_ + a1_;
    float sq  = a0_ * a0_ + a1_ * a1_;
    const float rk0 = r_k[h * HS + lane * 2];
    const float rk1 = r_k[h * HS + lane * 2 + 1];
    float dot = r[idx] * k[idx] * rk0 + r[idx + 1] * k[idx + 1] * rk1;
#pragma unroll
    for (int m = 16; m >= 1; m >>= 1) {
        sum += __shfl_xor(sum, m, 32);
        sq  += __shfl_xor(sq,  m, 32);
        dot += __shfl_xor(dot, m, 32);
    }
    const float mean = sum * (1.0f / HS);
    const float var  = sq * (1.0f / HS) - mean * mean;
    const float inv  = rsqrtf(var + 1e-5f * HS);

    const float o0 = (a0_ - mean) * inv * ln_w[c0]     + ln_b[c0]     + dot * v[idx];
    const float o1 = (a1_ - mean) * inv * ln_w[c0 + 1] + ln_b[c0 + 1] + dot * v[idx + 1];
    gated[idx]     = (bf16_t)(o0 * g[idx]);
    gated[idx + 1] = (bf16_t)(o1 * g[idx + 1]);
}

// ---------------------------------------------------------------------------
extern "C" void kernel_launch(void* const* d_in, const int* in_sizes, int n_in,
                              void* d_out, int out_size, void* d_ws, size_t ws_size,
                              hipStream_t stream)
{
    const float* x        = (const float*)d_in[0];
    const float* v_first  = (const float*)d_in[1];
    const float* shift    = (const float*)d_in[2];
    const float* wkv0     = (const float*)d_in[3];
    const float* x_r      = (const float*)d_in[4];
    const float* x_w      = (const float*)d_in[5];
    const float* x_k      = (const float*)d_in[6];
    const float* x_v      = (const float*)d_in[7];
    const float* x_a      = (const float*)d_in[8];
    const float* x_g      = (const float*)d_in[9];
    const float* w0       = (const float*)d_in[10];
    const float* w1       = (const float*)d_in[11];
    const float* w2       = (const float*)d_in[12];
    const float* a0       = (const float*)d_in[13];
    const float* a1       = (const float*)d_in[14];
    const float* a2       = (const float*)d_in[15];
    const float* v0       = (const float*)d_in[16];
    const float* v1       = (const float*)d_in[17];
    const float* v2       = (const float*)d_in[18];
    const float* g1       = (const float*)d_in[19];
    const float* g2       = (const float*)d_in[20];
    const float* k_k      = (const float*)d_in[21];
    const float* k_a      = (const float*)d_in[22];
    const float* r_k      = (const float*)d_in[23];
    const float* Wr       = (const float*)d_in[24];
    const float* Wk       = (const float*)d_in[25];
    const float* Wv       = (const float*)d_in[26];
    const float* Wo       = (const float*)d_in[27];
    const float* ln_w     = (const float*)d_in[28];
    const float* ln_b     = (const float*)d_in[29];

    // ---- workspace carve-out ----
    char* wsp = (char*)d_ws;
    size_t off = 0;
    auto alloc = [&](size_t bytes) -> void* {
        off = (off + 255) & ~(size_t)255;
        void* p = wsp + off;
        off += bytes;
        return p;
    };
    const size_t BTC = (size_t)NBT * CC;

    // bf16 staged activations (xr reused as 'gated' at the end)
    bf16_t* bxr = (bf16_t*)alloc(BTC * 2);
    bf16_t* bxw = (bf16_t*)alloc(BTC * 2);
    bf16_t* bxk = (bf16_t*)alloc(BTC * 2);
    bf16_t* bxv = (bf16_t*)alloc(BTC * 2);
    bf16_t* bxa = (bf16_t*)alloc(BTC * 2);
    bf16_t* bxg = (bf16_t*)alloc(BTC * 2);
    // bf16 weights (already [n,k] for the big GEMMs)
    bf16_t* bWr = (bf16_t*)alloc((size_t)CC * CC * 2);
    bf16_t* bWk = (bf16_t*)alloc((size_t)CC * CC * 2);
    bf16_t* bWv = (bf16_t*)alloc((size_t)CC * CC * 2);
    bf16_t* bWo = (bf16_t*)alloc((size_t)CC * CC * 2);
    // LoRA weights, transposed to [n,k] bf16
    bf16_t* w1t = (bf16_t*)alloc((size_t)96  * CC * 2);
    bf16_t* w2t = (bf16_t*)alloc((size_t)CC  * 96 * 2);
    bf16_t* a1t = (bf16_t*)alloc((size_t)96  * CC * 2);
    bf16_t* a2t = (bf16_t*)alloc((size_t)CC  * 96 * 2);
    bf16_t* v1t = (bf16_t*)alloc((size_t)64  * CC * 2);
    bf16_t* v2t = (bf16_t*)alloc((size_t)CC  * 64 * 2);
    bf16_t* g1t = (bf16_t*)alloc((size_t)256 * CC * 2);
    bf16_t* g2t = (bf16_t*)alloc((size_t)CC  * 256 * 2);
    // fp32 intermediates
    float* rbuf  = (float*)alloc(BTC * 4);
    float* kbuf  = (float*)alloc(BTC * 4);
    float* vbuf  = (float*)alloc(BTC * 4);
    float* wbuf  = (float*)alloc(BTC * 4);   // lora_w -> w
    float* abuf  = (float*)alloc(BTC * 4);   // lora_a -> iclr
    float* vlbuf = (float*)alloc(BTC * 4);   // lora_v -> kk
    float* gbuf  = (float*)alloc(BTC * 4);   // gate
    float* attb  = (float*)alloc(BTC * 4);
    // LoRA hidden (fp32) + bf16 activated
    float*  hw  = (float*)alloc((size_t)NBT * 96 * 4);
    float*  ha  = (float*)alloc((size_t)NBT * 96 * 4);
    float*  hv  = (float*)alloc((size_t)NBT * 64 * 4);
    float*  hg  = (float*)alloc((size_t)NBT * 256 * 4);
    bf16_t* hwb = (bf16_t*)alloc((size_t)NBT * 96 * 2);
    bf16_t* hab = (bf16_t*)alloc((size_t)NBT * 96 * 2);
    bf16_t* hvb = (bf16_t*)alloc((size_t)NBT * 64 * 2);
    bf16_t* hgb = (bf16_t*)alloc((size_t)NBT * 256 * 2);
    (void)ws_size; (void)n_in; (void)in_sizes; (void)out_size;

    const int TPB = 256;
    auto blocks = [](size_t n, int t) { return (unsigned)((n + t - 1) / t); };

    // 1) weight converts
    act_cvt<<<blocks((size_t)CC * CC, TPB), TPB, 0, stream>>>(Wr, bWr, (size_t)CC * CC, 0);
    act_cvt<<<blocks((size_t)CC * CC, TPB), TPB, 0, stream>>>(Wk, bWk, (size_t)CC * CC, 0);
    act_cvt<<<blocks((size_t)CC * CC, TPB), TPB, 0, stream>>>(Wv, bWv, (size_t)CC * CC, 0);
    act_cvt<<<blocks((size_t)CC * CC, TPB), TPB, 0, stream>>>(Wo, bWo, (size_t)CC * CC, 0);
    cvt_t<<<blocks((size_t)CC * 96,  TPB), TPB, 0, stream>>>(w1, w1t, CC, 96);
    cvt_t<<<blocks((size_t)96 * CC,  TPB), TPB, 0, stream>>>(w2, w2t, 96, CC);
    cvt_t<<<blocks((size_t)CC * 96,  TPB), TPB, 0, stream>>>(a1, a1t, CC, 96);
    cvt_t<<<blocks((size_t)96 * CC,  TPB), TPB, 0, stream>>>(a2, a2t, 96, CC);
    cvt_t<<<blocks((size_t)CC * 64,  TPB), TPB, 0, stream>>>(v1, v1t, CC, 64);
    cvt_t<<<blocks((size_t)64 * CC,  TPB), TPB, 0, stream>>>(v2, v2t, 64, CC);
    cvt_t<<<blocks((size_t)CC * 256, TPB), TPB, 0, stream>>>(g1, g1t, CC, 256);
    cvt_t<<<blocks((size_t)256 * CC, TPB), TPB, 0, stream>>>(g2, g2t, 256, CC);

    // 2) token shift -> bf16 staged activations
    prep_x<<<blocks(BTC, TPB), TPB, 0, stream>>>(x, shift, x_r, x_w, x_k, x_v, x_a, x_g,
                                                 bxr, bxw, bxk, bxv, bxa, bxg);

    // 3) big GEMMs: r, k, v  (block tile 256x128)
    dim3 gBig((CC + 127) / 128, NBT / 256);
    gemm_bf16_nt<<<gBig, TPB, 0, stream>>>(bxr, bWr, rbuf, NBT, CC, CC);
    gemm_bf16_nt<<<gBig, TPB, 0, stream>>>(bxk, bWk, kbuf, NBT, CC, CC);
    gemm_bf16_nt<<<gBig, TPB, 0, stream>>>(bxv, bWv, vbuf, NBT, CC, CC);

    // 4) LoRA down-projections
    gemm_bf16_nt<<<dim3(1, NBT / 256), TPB, 0, stream>>>(bxw, w1t, hw, NBT, 96,  CC);
    gemm_bf16_nt<<<dim3(1, NBT / 256), TPB, 0, stream>>>(bxa, a1t, ha, NBT, 96,  CC);
    gemm_bf16_nt<<<dim3(1, NBT / 256), TPB, 0, stream>>>(bxv, v1t, hv, NBT, 64,  CC);
    gemm_bf16_nt<<<dim3(2, NBT / 256), TPB, 0, stream>>>(bxg, g1t, hg, NBT, 256, CC);

    // 5) hidden activations -> bf16
    act_cvt<<<blocks((size_t)NBT * 96,  TPB), TPB, 0, stream>>>(hw, hwb, (size_t)NBT * 96, 1);
    act_cvt<<<blocks((size_t)NBT * 96,  TPB), TPB, 0, stream>>>(ha, hab, (size_t)NBT * 96, 0);
    act_cvt<<<blocks((size_t)NBT * 64,  TPB), TPB, 0, stream>>>(hv, hvb, (size_t)NBT * 64, 0);
    act_cvt<<<blocks((size_t)NBT * 256, TPB), TPB, 0, stream>>>(hg, hgb, (size_t)NBT * 256, 2);

    // 6) LoRA up-projections
    gemm_bf16_nt<<<gBig, TPB, 0, stream>>>(hwb, w2t, wbuf,  NBT, CC, 96);
    gemm_bf16_nt<<<gBig, TPB, 0, stream>>>(hab, a2t, abuf,  NBT, CC, 96);
    gemm_bf16_nt<<<gBig, TPB, 0, stream>>>(hvb, v2t, vlbuf, NBT, CC, 64);
    gemm_bf16_nt<<<gBig, TPB, 0, stream>>>(hgb, g2t, gbuf,  NBT, CC, 256);

    // 7) post-mix elementwise + per-head kk normalization
    post_mix<<<(unsigned)((size_t)NBT * HEADS / 8), TPB, 0, stream>>>(
        kbuf, vbuf, wbuf, abuf, vlbuf, v_first, w0, a0, v0, k_k, k_a);

    // 8) sequential scan over T
    rwkv_scan<<<BB * HEADS, HS, 0, stream>>>(rbuf, wbuf, kbuf, vbuf, vlbuf, abuf,
                                             wkv0, attb);

    // 9) GroupNorm + bonus + gate -> bf16 (reuse bxr)
    gn_bonus<<<(unsigned)((size_t)NBT * HEADS / 8), TPB, 0, stream>>>(
        attb, rbuf, kbuf, vbuf, gbuf, ln_w, ln_b, r_k, bxr);

    // 10) output projection
    gemm_bf16_nt<<<gBig, TPB, 0, stream>>>(bxr, bWo, (float*)d_out, NBT, CC, CC);
}
